// DecoderBlock_46755013984985
// MI455X (gfx1250) — matrix operand order repaired and checked by reading, compiled, not run on hardware
//
#include <hip/hip_runtime.h>

// ---------------------------------------------------------------------------
// Types for CDNA5 WMMA (wave32)
// ---------------------------------------------------------------------------
typedef __bf16 bf16_t;
typedef bf16_t   v16bf __attribute__((ext_vector_type(16)));
typedef float    v8f   __attribute__((ext_vector_type(8)));
typedef unsigned v8u   __attribute__((ext_vector_type(8)));

#define BATCH 16
#define MPTS  1024
#define NPTS  4096
#define DFEAT 256
#define CFEAT 128
#define DIN   384          // CFEAT + DFEAT
#define DOUT  256
#define NB    (BATCH * NPTS)   // 65536 rows for the MLP GEMMs

__device__ __forceinline__ unsigned short f32_to_bf16(float f) {
    unsigned u = __float_as_uint(f);
    unsigned r = u + 0x7FFFu + ((u >> 16) & 1u);   // round-to-nearest-even
    return (unsigned short)(r >> 16);
}
__device__ __forceinline__ float bf16us_to_f32(unsigned short u) {
    return __uint_as_float(((unsigned)u) << 16);
}
__device__ __forceinline__ float fastrcp(float x) {   // v_rcp_f32
    return __builtin_amdgcn_rcpf(x);
}
__device__ __forceinline__ float fastrsq(float x) {   // v_rsq_f32
    return __builtin_amdgcn_rsqf(x);
}
__device__ __forceinline__ v16bf pack_frag(v8u raw) {
    union { v8u u; v16bf b; } c;
    c.u = raw;
    return c.b;
}

__device__ __forceinline__ v8f wmma_bf16(v16bf a, v16bf b, v8f c) {
    // 8 args: (neg_a, A, neg_b, B, c_mod, C, reuse_a, reuse_b)
    return __builtin_amdgcn_wmma_f32_16x16x32_bf16(false, a, false, b,
                                                   (short)0, c, false, false);
}

// ---------------------------------------------------------------------------
// Async global -> LDS copy (CDNA5 GLOBAL_LOAD_ASYNC_TO_LDS_B32, ASYNCcnt).
// LDS destination VGPR holds the LDS byte offset: the flat LDS aperture keeps
// the offset in addr[31:0], so truncating the generic pointer recovers it.
// ---------------------------------------------------------------------------
__device__ __forceinline__ void async_g2lds_b32(const void* gptr, void* lptr) {
    unsigned lds = (unsigned)(unsigned long long)lptr;
    asm volatile("global_load_async_to_lds_b32 %0, %1, off"
                 :: "v"(lds), "v"(gptr) : "memory");
}
__device__ __forceinline__ void wait_async0() {
    asm volatile("s_wait_asynccnt 0" ::: "memory");
}

// ---------------------------------------------------------------------------
// Fragment loaders (CDNA5 ISA 7.12.2 layouts). Each dword read holds the
// (even,odd) bf16 pair in (lo,hi) halves -- exactly the WMMA register image.
// ---------------------------------------------------------------------------
// A: 16x32 bf16, LDS tile row-major [row][k], stride even.
// lane 0-15 : M = lane,     K in {0..7} U {16..23}
// lane 16-31: M = lane-16,  K in {8..15} U {24..31}
__device__ __forceinline__ v16bf frag_a_rm(const unsigned short* tile,
                                           int row0, int stride, int lane) {
    const int half = lane >> 4;
    const int r    = lane & 15;
    const unsigned short* rp = tile + (row0 + r) * stride;
    v8u raw;
#pragma unroll
    for (int v = 0; v < 8; ++v) {
        int kb = (v < 4) ? (8 * half + 2 * v) : (16 + 8 * half + 2 * (v - 4));
        raw[v] = *(const unsigned*)&rp[kb];
    }
    return pack_frag(raw);
}

// B: 32x16 bf16, LDS tile n-major [n][k] (stride even).
// lane 0-15 : N = lane,     K in {0..7} U {16..23}
// lane 16-31: N = lane-16,  K in {8..15} U {24..31}
__device__ __forceinline__ v16bf frag_b_nm(const unsigned short* tile,
                                           int n0, int stride, int lane) {
    const int half = lane >> 4;
    const int n    = lane & 15;
    const unsigned short* rp = tile + (n0 + n) * stride;
    v8u raw;
#pragma unroll
    for (int v = 0; v < 8; ++v) {
        int kb = (v < 4) ? (8 * half + 2 * v) : (16 + 8 * half + 2 * (v - 4));
        raw[v] = *(const unsigned*)&rp[kb];
    }
    return pack_frag(raw);
}

// ---------------------------------------------------------------------------
// Prep kernels
// ---------------------------------------------------------------------------
__global__ void k_f32_to_bf16(const float* __restrict__ in,
                              unsigned short* __restrict__ out, int n) {
    for (int i = blockIdx.x * blockDim.x + threadIdx.x; i < n;
         i += gridDim.x * blockDim.x)
        out[i] = f32_to_bf16(in[i]);
}

// feat_up (B*N, 128) -> x_bf[p*384 + c], c in [0,128)
__global__ void k_pack_featup(const float* __restrict__ fu,
                              unsigned short* __restrict__ xbf) {
    const int total = NB * CFEAT;
    for (int i = blockIdx.x * blockDim.x + threadIdx.x; i < total;
         i += gridDim.x * blockDim.x) {
        int p = i >> 7, c = i & 127;
        xbf[p * DIN + c] = f32_to_bf16(fu[i]);
    }
}

// ---------------------------------------------------------------------------
// Fused inverse-distance interpolation GEMM:
//   x[:, :, 128:384] = normalize(1/(d^2+eps)) @ feat_down       (bf16 out)
// Block: 64 N-rows x 256 D-cols, 8 waves (2 row x 4 col), K=M=1024 in 32 tiles
// ---------------------------------------------------------------------------
__global__ __launch_bounds__(256)
void k_interp(const float* __restrict__ xyz_down, const float* __restrict__ xyz_up,
              const unsigned short* __restrict__ fd_bf,
              unsigned short* __restrict__ x_bf) {
    __shared__ __align__(16) float dn_s[MPTS * 3];        // 12 KB
    __shared__ __align__(16) float up_s[64 * 3];
    __shared__ float rs_s[64];
    __shared__ __align__(16) unsigned short At[64 * 36];   // weights, row-major
    __shared__ __align__(16) unsigned short Bt[256 * 36];  // feat_down, n-major

    const int t     = threadIdx.x;
    const int b     = blockIdx.y;
    const int nbase = blockIdx.x * 64;
    const int lane  = t & 31, wave = t >> 5;
    const int wr    = wave >> 2, wc = wave & 3;

    // ---- stage 0: xyz into LDS (async global->LDS for the big copy) ------
    for (int e = t; e < MPTS * 3; e += 256)
        async_g2lds_b32(&xyz_down[b * MPTS * 3 + e], &dn_s[e]);
    for (int e = t; e < 64 * 3;  e += 256)
        up_s[e] = xyz_up[(b * NPTS + nbase) * 3 + e];
    if (t < 64) rs_s[t] = 0.0f;
    wait_async0();
    __syncthreads();

    // ---- stage 1: per-row reciprocal-distance sums -----------------------
    {
        const int r = t >> 2, q = t & 3;
        const float ux = up_s[r * 3], uy = up_s[r * 3 + 1], uz = up_s[r * 3 + 2];
        float s = 0.0f;
        for (int m = q; m < MPTS; m += 4) {
            float dx = ux - dn_s[m * 3];
            float dy = uy - dn_s[m * 3 + 1];
            float dz = uz - dn_s[m * 3 + 2];
            s += fastrcp(dx * dx + dy * dy + dz * dz + 1e-8f);
        }
        atomicAdd(&rs_s[r], s);
    }
    __syncthreads();
    if (t < 64) rs_s[t] = fastrcp(rs_s[t]);
    __syncthreads();

    // ---- stage 2: K loop -------------------------------------------------
    v8f acc[2][4];
    const v8f vz = {0.f, 0.f, 0.f, 0.f, 0.f, 0.f, 0.f, 0.f};
#pragma unroll
    for (int i = 0; i < 2; ++i)
#pragma unroll
        for (int j = 0; j < 4; ++j) acc[i][j] = vz;

    for (int kt = 0; kt < 32; ++kt) {
        // regenerate 64x32 normalized weight tile (bf16 pairs, dword stores)
        for (int e = t; e < 1024; e += 256) {
            int r = e >> 4, kk = (e & 15) * 2, m = kt * 32 + kk;
            float ux = up_s[r * 3], uy = up_s[r * 3 + 1], uz = up_s[r * 3 + 2];
            float inv = rs_s[r];
            float dx0 = ux - dn_s[m * 3],     dy0 = uy - dn_s[m * 3 + 1],
                  dz0 = uz - dn_s[m * 3 + 2];
            float dx1 = ux - dn_s[m * 3 + 3], dy1 = uy - dn_s[m * 3 + 4],
                  dz1 = uz - dn_s[m * 3 + 5];
            float w0 = fastrcp(dx0 * dx0 + dy0 * dy0 + dz0 * dz0 + 1e-8f) * inv;
            float w1 = fastrcp(dx1 * dx1 + dy1 * dy1 + dz1 * dz1 + 1e-8f) * inv;
            *(unsigned*)&At[r * 36 + kk] =
                (unsigned)f32_to_bf16(w0) | ((unsigned)f32_to_bf16(w1) << 16);
        }
        // stream 32x256 feat_down tile -> n-major LDS (transpose on store)
        for (int e = t; e < 4096; e += 256) {
            int kk = e >> 7, d2 = (e & 127) * 2;
            unsigned pr = *(const unsigned*)
                &fd_bf[(size_t)((b << 10) + kt * 32 + kk) * DFEAT + d2];
            Bt[(d2    ) * 36 + kk] = (unsigned short)(pr & 0xFFFFu);
            Bt[(d2 + 1) * 36 + kk] = (unsigned short)(pr >> 16);
        }
        __syncthreads();

        v16bf af0 = frag_a_rm(At, wr * 32,      36, lane);
        v16bf af1 = frag_a_rm(At, wr * 32 + 16, 36, lane);
#pragma unroll
        for (int ct = 0; ct < 4; ++ct) {
            v16bf bf = frag_b_nm(Bt, wc * 64 + ct * 16, 36, lane);
            acc[0][ct] = wmma_bf16(af0, bf, acc[0][ct]);
            acc[1][ct] = wmma_bf16(af1, bf, acc[1][ct]);
        }
        __syncthreads();
    }

    // ---- epilogue: bf16 into x[:, :, 128 + d] ----------------------------
    const int half = lane >> 4, ln = lane & 15;
#pragma unroll
    for (int rt = 0; rt < 2; ++rt) {
        int rowb = nbase + wr * 32 + rt * 16 + half * 8;
#pragma unroll
        for (int ct = 0; ct < 4; ++ct) {
            int col = wc * 64 + ct * 16 + ln;
#pragma unroll
            for (int v = 0; v < 8; ++v) {
                int n = rowb + v;
                x_bf[((size_t)(b << 12) + n) * DIN + CFEAT + col] =
                    f32_to_bf16(acc[rt][ct][v]);
            }
        }
    }
}

// ---------------------------------------------------------------------------
// GEMM: out(p,col) = sum_k A(p,k) * W(col,k) + bias(col), fused BN statistics.
// A: (NB x 384) bf16 row-major. W: (ncols_total x 384) bf16 row-major.
// Block tile 64x128, 8 waves (2x4), wave tile 32x32 (4 WMMA / k-step).
// Tile fills use GLOBAL_LOAD_ASYNC_TO_LDS_B32 (ASYNCcnt).
// ---------------------------------------------------------------------------
template <bool F32OUT>
__global__ __launch_bounds__(256)
void k_gemm(const unsigned short* __restrict__ Abf,
            const unsigned short* __restrict__ Wbf,
            const float* __restrict__ bias,
            unsigned short* __restrict__ obf, float* __restrict__ of32,
            float* __restrict__ gsum, float* __restrict__ gsq, int ncols) {
    __shared__ __align__(16) unsigned short At[64 * 36];
    __shared__ __align__(16) unsigned short Btl[128 * 36];

    const int t    = threadIdx.x;
    const int lane = t & 31, wave = t >> 5;
    const int wr   = wave >> 2, wc = wave & 3;
    const int rowb = blockIdx.x * 64;
    const int colb = blockIdx.y * 128;

    v8f acc[2][2];
    const v8f vz = {0.f, 0.f, 0.f, 0.f, 0.f, 0.f, 0.f, 0.f};
    acc[0][0] = vz; acc[0][1] = vz; acc[1][0] = vz; acc[1][1] = vz;

    for (int k0 = 0; k0 < DIN; k0 += 32) {
        // A: 64x32 bf16 and B: 128x32 bf16, async dword copies into LDS
#pragma unroll
        for (int e = t; e < 1024; e += 256) {
            int r = e >> 4, kk = (e & 15) * 2;
            async_g2lds_b32(&Abf[(size_t)(rowb + r) * DIN + k0 + kk],
                            &At[r * 36 + kk]);
        }
#pragma unroll
        for (int e = t; e < 2048; e += 256) {
            int n = e >> 4, kk = (e & 15) * 2;
            async_g2lds_b32(&Wbf[(size_t)(colb + n) * DIN + k0 + kk],
                            &Btl[n * 36 + kk]);
        }
        wait_async0();
        __syncthreads();

        v16bf a0 = frag_a_rm(At, wr * 32,      36, lane);
        v16bf a1 = frag_a_rm(At, wr * 32 + 16, 36, lane);
        v16bf b0 = frag_b_nm(Btl, wc * 32,      36, lane);
        v16bf b1 = frag_b_nm(Btl, wc * 32 + 16, 36, lane);
        acc[0][0] = wmma_bf16(a0, b0, acc[0][0]);
        acc[0][1] = wmma_bf16(a0, b1, acc[0][1]);
        acc[1][0] = wmma_bf16(a1, b0, acc[1][0]);
        acc[1][1] = wmma_bf16(a1, b1, acc[1][1]);
        __syncthreads();
    }

    // epilogue: bias, store, per-channel sum / sumsq for train-mode BN
    const int half = lane >> 4, ln = lane & 15;
#pragma unroll
    for (int ct = 0; ct < 2; ++ct) {
        const int colg = colb + wc * 32 + ct * 16 + ln;
        const float bv = bias[colg];
        float s = 0.0f, ss = 0.0f;
#pragma unroll
        for (int rt = 0; rt < 2; ++rt) {
            int rbase = rowb + wr * 32 + rt * 16 + half * 8;
#pragma unroll
            for (int v = 0; v < 8; ++v) {
                float val = acc[rt][ct][v] + bv;
                size_t idx = (size_t)(rbase + v) * ncols + colg;
                if (F32OUT) of32[idx] = val;
                else        obf[idx]  = f32_to_bf16(val);
                s += val; ss += val * val;
            }
        }
        s  += __shfl_xor(s, 16, 32);
        ss += __shfl_xor(ss, 16, 32);
        if (lane < 16) {
            atomicAdd(&gsum[colg], s);
            atomicAdd(&gsq[colg], ss);
        }
    }
}

// ---------------------------------------------------------------------------
// BatchNorm finalize kernels (train-mode, biased variance)
// ---------------------------------------------------------------------------
__global__ void k_bn_relu_bf(unsigned short* __restrict__ h,
                             const float* __restrict__ sum, const float* __restrict__ sq,
                             const float* __restrict__ gamma, const float* __restrict__ beta,
                             int ncols, int total, float invN) {
    for (int i = blockIdx.x * blockDim.x + threadIdx.x; i < total;
         i += gridDim.x * blockDim.x) {
        int col = i % ncols;
        float mu  = sum[col] * invN;
        float var = fmaxf(sq[col] * invN - mu * mu, 0.0f);
        float sc  = gamma[col] * fastrsq(var + 1e-5f);
        float v   = (bf16us_to_f32(h[i]) - mu) * sc + beta[col];
        h[i] = f32_to_bf16(fmaxf(v, 0.0f));
    }
}

__global__ void k_bn_f32(float* __restrict__ h,
                         const float* __restrict__ sum, const float* __restrict__ sq,
                         const float* __restrict__ gamma, const float* __restrict__ beta,
                         int ncols, int total, float invN) {
    for (int i = blockIdx.x * blockDim.x + threadIdx.x; i < total;
         i += gridDim.x * blockDim.x) {
        int col = i % ncols;
        float mu  = sum[col] * invN;
        float var = fmaxf(sq[col] * invN - mu * mu, 0.0f);
        float sc  = gamma[col] * fastrsq(var + 1e-5f);
        h[i] = (h[i] - mu) * sc + beta[col];
    }
}

// ---------------------------------------------------------------------------
// Host-side orchestration
// ---------------------------------------------------------------------------
extern "C" void kernel_launch(void* const* d_in, const int* in_sizes, int n_in,
                              void* d_out, int out_size, void* d_ws, size_t ws_size,
                              hipStream_t stream) {
    const float* xyz_down  = (const float*)d_in[0];
    const float* xyz_up    = (const float*)d_in[1];
    const float* feat_down = (const float*)d_in[2];
    const float* feat_up   = (const float*)d_in[3];
    const float* W1  = (const float*)d_in[4];
    const float* b1  = (const float*)d_in[5];
    const float* g1  = (const float*)d_in[6];
    const float* be1 = (const float*)d_in[7];
    const float* W2  = (const float*)d_in[8];
    const float* b2  = (const float*)d_in[9];
    const float* g2  = (const float*)d_in[10];
    const float* be2 = (const float*)d_in[11];
    float* out = (float*)d_out;
    char*  ws  = (char*)d_ws;

    size_t off = 0;
    auto take = [&](size_t bytes) {
        size_t c = off;
        off += (bytes + 255) & ~(size_t)255;
        return c;
    };
    unsigned short* fd_bf = (unsigned short*)(ws + take((size_t)BATCH * MPTS * DFEAT * 2));
    unsigned short* w1_bf = (unsigned short*)(ws + take((size_t)DIN * DIN * 2));
    unsigned short* w2_bf = (unsigned short*)(ws + take((size_t)DOUT * DIN * 2));
    unsigned short* x_bf  = (unsigned short*)(ws + take((size_t)NB * DIN * 2));
    unsigned short* h1_bf = (unsigned short*)(ws + take((size_t)NB * DIN * 2));
    float* stats = (float*)(ws + take(1280 * 4));
    float* sum1 = stats;        float* sq1 = stats + 384;
    float* sum2 = stats + 768;  float* sq2 = stats + 1024;

    // zero BN accumulators (memset node is graph-capturable)
    hipMemsetAsync(stats, 0, 1280 * 4, stream);

    // precision conversion / packing
    k_f32_to_bf16<<<dim3(2048), 256, 0, stream>>>(feat_down, fd_bf, BATCH * MPTS * DFEAT);
    k_f32_to_bf16<<<dim3(288),  256, 0, stream>>>(W1, w1_bf, DIN * DIN);
    k_f32_to_bf16<<<dim3(192),  256, 0, stream>>>(W2, w2_bf, DOUT * DIN);
    k_pack_featup<<<dim3(4096), 256, 0, stream>>>(feat_up, x_bf);

    // fused distance-weighted interpolation GEMM (writes x[:, :, 128:384])
    k_interp<<<dim3(NPTS / 64, BATCH), 256, 0, stream>>>(xyz_down, xyz_up, fd_bf, x_bf);

    // MLP layer 1: (65536x384)x(384x384) + b1, accumulate BN1 stats
    k_gemm<false><<<dim3(NB / 64, DIN / 128), 256, 0, stream>>>(
        x_bf, w1_bf, b1, h1_bf, nullptr, sum1, sq1, DIN);
    k_bn_relu_bf<<<dim3(8192), 256, 0, stream>>>(h1_bf, sum1, sq1, g1, be1,
                                                 DIN, NB * DIN, 1.0f / NB);

    // MLP layer 2: (65536x384)x(384x256) + b2 -> d_out (f32), BN2 stats
    k_gemm<true><<<dim3(NB / 64, DOUT / 128), 256, 0, stream>>>(
        h1_bf, w2_bf, b2, nullptr, out, sum2, sq2, DOUT);
    k_bn_f32<<<dim3(8192), 256, 0, stream>>>(out, sum2, sq2, g2, be2,
                                             DOUT, NB * DOUT, 1.0f / NB);
}
